// RNN_81810537054685
// MI455X (gfx1250) — matrix operand order repaired and checked by reading, not compiled, not run on hardware
//
#include <hip/hip_runtime.h>
#include <math.h>

typedef unsigned short u16;
typedef __bf16 bf16_t;
typedef bf16_t bfrag __attribute__((ext_vector_type(16)));
typedef float f32x8 __attribute__((ext_vector_type(8)));

union FragU {
    bfrag v;
    uint4 q[2];
    u16   u[16];
};

// native hardware bf16 convert (round-to-nearest-even)
__device__ __forceinline__ u16 f2bf(float x) {
    bf16_t h = (bf16_t)x;
    return __builtin_bit_cast(u16, h);
}
__device__ __forceinline__ float bf2f(u16 h) {
    unsigned u = ((unsigned)h) << 16;
    return __builtin_bit_cast(float, u);
}

// fast tanh: one v_exp_f32 (TRANS, co-executes with VALU) + rcp-based divide
__device__ __forceinline__ float fast_tanh(float x) {
    x = fminf(fmaxf(x, -15.0f), 15.0f);   // saturate; avoids inf/inf
    float e = __expf(2.0f * x);
    return __fdividef(e - 1.0f, e + 1.0f);
}

// ---------------------------------------------------------------------------
// Async global -> LDS copy (CDNA5, tracked by ASYNCcnt)
// ---------------------------------------------------------------------------
typedef __attribute__((address_space(3))) void lds_void_t;

__device__ __forceinline__ unsigned lds_addr_of(void* p) {
    return (unsigned)(unsigned long long)(lds_void_t*)p;
}

__device__ __forceinline__ void async_copy_b128(void* lds_ptr, const void* gptr) {
    unsigned l = lds_addr_of(lds_ptr);
    asm volatile("global_load_async_to_lds_b128 %0, %1, off"
                 :: "v"(l), "v"(gptr)
                 : "memory");
}

__device__ __forceinline__ void wait_asynccnt0() {
    asm volatile("s_wait_asynccnt 0x0" ::: "memory");
}

// ---------------------------------------------------------------------------
// f32 -> bf16 conversion, 8 elements / thread (pure bandwidth pass)
// ---------------------------------------------------------------------------
__global__ void f32_to_bf16_vec8(const float* __restrict__ in,
                                 u16* __restrict__ out, long long n8) {
    long long i = (long long)blockIdx.x * blockDim.x + threadIdx.x;
    if (i >= n8) return;
    const float4* p = (const float4*)in + i * 2;
    float4 a = p[0];
    float4 b = p[1];
    union { u16 u[8]; uint4 q; } r;
    r.u[0] = f2bf(a.x); r.u[1] = f2bf(a.y); r.u[2] = f2bf(a.z); r.u[3] = f2bf(a.w);
    r.u[4] = f2bf(b.x); r.u[5] = f2bf(b.y); r.u[6] = f2bf(b.z); r.u[7] = f2bf(b.w);
    ((uint4*)out)[i] = r.q;
}

// ---------------------------------------------------------------------------
// WMMA GEMM:  C[M,N] = act( A[M,K] @ W[N,K]^T + bias + addend )
//   A : bf16 row-major (staged per-k-chunk into LDS via async copies)
//   W : bf16 [N,K] row-major (torch [out,in]) -> each row is one B-column
//   OUT == 0 : store bf16        OUT == 1 : store f32 with fast_tanh
// All strides / K are compile-time so element addressing folds into the
// 24-bit instruction offset field (one base address per tile).
// ---------------------------------------------------------------------------
template <int WM, int WN, int MT, int NT, int OUT,
          int LDA, int LDW, int LDADD, int LDC, int K>
__global__ __launch_bounds__(WM * WN * 32)
void wmma_gemm_kernel(const u16* __restrict__ A,
                      const u16* __restrict__ W,
                      const float* __restrict__ bias,
                      const u16* __restrict__ addend,
                      void* __restrict__ Cout) {
    constexpr int NTHREADS = WM * WN * 32;
    constexpr int BM = WM * MT * 16;            // block rows
    constexpr int CHUNKS = (BM * 4) / NTHREADS; // 16B chunks per thread
    static_assert((BM * 4) % NTHREADS == 0, "tile/thread mismatch");

    __shared__ __align__(16) u16 lds_a[BM * 32];   // BM x 32 bf16 A chunk

    const int lane = threadIdx.x & 31;
    const int wave = threadIdx.x >> 5;
    const int hl   = lane >> 4;     // half-wave: K-group (A,B) / M-group (C)
    const int lc   = lane & 15;     // row (A) / column (B,C)
    const int wm   = wave % WM;
    const int wn   = wave / WM;
    const int block_m = blockIdx.x * BM;
    const int m_base  = block_m + wm * MT * 16;
    const int n_base  = blockIdx.y * (WN * NT * 16) + wn * NT * 16;

    f32x8 acc[MT][NT] = {};

    if (A != nullptr) {
        for (int k0 = 0; k0 < K; k0 += 32) {
            // ---- stage A block-tile (BM x 32) into LDS, async ----
#pragma unroll
            for (int c = 0; c < CHUNKS; ++c) {
                const int idx = c * NTHREADS + threadIdx.x; // 16B chunk id
                const int row = idx >> 2;                   // 4 chunks per row
                const int col = (idx & 3) * 8;              // element offset
                const u16* g = A + (size_t)(block_m + row) * LDA + k0 + col;
                async_copy_b128(&lds_a[idx * 8], g);
            }
            wait_asynccnt0();
            __syncthreads();

            // ---- fragments ----
            bfrag a[MT];
            bfrag b[NT];
#pragma unroll
            for (int i = 0; i < MT; ++i) {
                // elems 0..7 <- k-local hl*8, elems 8..15 <- +16
                const u16* ab = &lds_a[(wm * MT * 16 + i * 16 + lc) * 32 + hl * 8];
                FragU f;
                f.q[0] = *(const uint4*)(ab);
                f.q[1] = *(const uint4*)(ab + 16);
                a[i] = f.v;
            }
#pragma unroll
            for (int j = 0; j < NT; ++j) {
                // lane lc holds B-column (n_base + j*16 + lc): elems 0..7 <- k-local hl*8, elems 8..15 <- +16
                const u16* wb = W + (size_t)(n_base + j * 16 + lc) * LDW +
                                k0 + hl * 8;
                FragU f;
                f.q[0] = *(const uint4*)(wb);
                f.q[1] = *(const uint4*)(wb + 16);
                b[j] = f.v;
            }
            // keep the weight stream warm (L2 / WGP$)
            __builtin_prefetch(
                (const void*)(W + (size_t)(n_base + lc) * LDW + k0 + 32), 0, 3);

#pragma unroll
            for (int i = 0; i < MT; ++i)
#pragma unroll
                for (int j = 0; j < NT; ++j)
                    acc[i][j] = __builtin_amdgcn_wmma_f32_16x16x32_bf16(
                        false, a[i], false, b[j], (short)0, acc[i][j],
                        false, false);

            __syncthreads();   // protect LDS tile before next-iter overwrite
        }
    }

    // Epilogue: C/D layout -> VGPR v : M = v + hl*8, col = lc
#pragma unroll
    for (int i = 0; i < MT; ++i) {
#pragma unroll
        for (int j = 0; j < NT; ++j) {
            const int n = n_base + j * 16 + lc;
            const int m0 = m_base + i * 16 + hl * 8;
            const float bv = (bias != nullptr) ? bias[n] : 0.0f;
            // per-tile bases; per-element offsets are compile-time immediates
            const u16*   addb = (addend != nullptr)
                                    ? addend + (size_t)m0 * LDADD + n : nullptr;
            u16*   cb16 = (u16*)Cout + (size_t)m0 * LDC + n;
            float* cb32 = (float*)Cout + (size_t)m0 * LDC + n;
#pragma unroll
            for (int v = 0; v < 8; ++v) {
                float val = acc[i][j][v] + bv;
                if (addend != nullptr)
                    val += bf2f(addb[(size_t)v * LDADD]);
                if (OUT == 0)
                    cb16[(size_t)v * LDC] = f2bf(val);
                else
                    cb32[(size_t)v * LDC] = fast_tanh(val);
            }
        }
    }
}

// ---------------------------------------------------------------------------
// Launch: convert -> GEMM1 (input proj) -> 512 serial step-GEMMs -> GEMM3
// ---------------------------------------------------------------------------
extern "C" void kernel_launch(void* const* d_in, const int* in_sizes, int n_in,
                              void* d_out, int out_size, void* d_ws,
                              size_t ws_size, hipStream_t stream) {
    constexpr int T = 512, B = 64, F = 1024, H = 2048;

    const float* x  = (const float*)d_in[0];   // [T,B,F]
    const float* Wx = (const float*)d_in[1];   // [H,F]
    const float* bx = (const float*)d_in[2];   // [H]
    const float* Wu = (const float*)d_in[3];   // [H,H]
    const float* bu = (const float*)d_in[4];   // [H]
    const float* Wo = (const float*)d_in[5];   // [F,H]
    const float* bo = (const float*)d_in[6];   // [F]
    float* out = (float*)d_out;                // [T,B,F]

    constexpr size_t nX  = (size_t)T * B * F;
    constexpr size_t nWx = (size_t)H * F;
    constexpr size_t nWu = (size_t)H * H;
    constexpr size_t nWo = (size_t)F * H;
    constexpr size_t nTBH = (size_t)T * B * H;

    u16* ws    = (u16*)d_ws;
    u16* x_bf  = ws;                  //  64 MB : x in bf16
    u16* Wx_bf = x_bf + nX;           //   4 MB
    u16* Wu_bf = Wx_bf + nWx;         //   8 MB (L2-resident during the scan)
    u16* Wo_bf = Wu_bf + nWu;         //   4 MB
    u16* WX    = Wo_bf + nWo;         // 128 MB : wx_out bf16 [T*B, H]
    u16* Hall  = WX + nTBH;           // 128 MB : all h_t bf16 [T*B, H]

    // 1) bf16 conversions (memory-bound; ~0.3 GB total traffic)
    f32_to_bf16_vec8<<<(int)((nX / 8 + 255) / 256), 256, 0, stream>>>(
        x, x_bf, (long long)(nX / 8));
    f32_to_bf16_vec8<<<(int)((nWx / 8 + 255) / 256), 256, 0, stream>>>(
        Wx, Wx_bf, (long long)(nWx / 8));
    f32_to_bf16_vec8<<<(int)((nWu / 8 + 255) / 256), 256, 0, stream>>>(
        Wu, Wu_bf, (long long)(nWu / 8));
    f32_to_bf16_vec8<<<(int)((nWo / 8 + 255) / 256), 256, 0, stream>>>(
        Wo, Wo_bf, (long long)(nWo / 8));

    // 2) GEMM1: WX = x @ Wx^T + bx   (M=32768, N=2048, K=1024)
    //    block tile 128x128 (4x2 waves, 32x64 per wave)
    wmma_gemm_kernel<4, 2, 2, 4, 0, /*LDA*/ F, /*LDW*/ F, /*LDADD*/ 1,
                     /*LDC*/ H, /*K*/ F>
        <<<dim3((T * B) / 128, H / 128), 256, 0, stream>>>(
            x_bf, Wx_bf, bx, nullptr, (void*)WX);

    // 3) serial scan: h_t = WX[t] + h_{t-1} @ Wu^T + bu
    //    (M=64, N=2048, K=2048); block tile 64x64 (4x2 waves, 16x32/wave)
    //    -> 32 blocks per step to spread the 8 MB L2-resident Wu stream.
    for (int t = 0; t < T; ++t) {
        const u16* Aprev = (t == 0) ? nullptr : Hall + (size_t)(t - 1) * B * H;
        wmma_gemm_kernel<4, 2, 1, 2, 0, /*LDA*/ H, /*LDW*/ H, /*LDADD*/ H,
                         /*LDC*/ H, /*K*/ H>
            <<<dim3(1, H / 64), 256, 0, stream>>>(
                Aprev, Wu_bf, bu, WX + (size_t)t * B * H,
                (void*)(Hall + (size_t)t * B * H));
    }

    // 4) GEMM3: out = tanh(Hall @ Wo^T + bo)  (M=32768, N=1024, K=2048)
    wmma_gemm_kernel<4, 2, 2, 4, 1, /*LDA*/ H, /*LDW*/ H, /*LDADD*/ 1,
                     /*LDC*/ F, /*K*/ H>
        <<<dim3((T * B) / 128, F / 128), 256, 0, stream>>>(
            Hall, Wo_bf, bo, nullptr, (void*)out);
}
